// LinearAndTranspose_84739704750321
// MI455X (gfx1250) — matrix operand order repaired and checked by reading, not compiled, not run on hardware
//
#include <hip/hip_runtime.h>

// Problem constants (from reference)
#define IN_F   5760
#define OUT_F  640
#define RANK   32
#define GROUP  64
#define NTOK   16384          // 4 * 4096 tokens
#define KB     (IN_F / 32)    // 180 k-blocks of 32 (one f16 WMMA each)
#define MTILES (NTOK / 16)    // 1024
#define NTILES (OUT_F / 16)   // 40
#define MT     2              // m-tiles per wave (register blocking)
#define NT     2              // n-tiles per wave

typedef __attribute__((ext_vector_type(16))) _Float16 v16h;
typedef __attribute__((ext_vector_type(8)))  float    v8f;
typedef __attribute__((ext_vector_type(2)))  float    v2f;
typedef __attribute__((ext_vector_type(4)))  _Float16 h4;
typedef __attribute__((ext_vector_type(8)))  _Float16 h8;

#if defined(__has_builtin)
#if __has_builtin(__builtin_amdgcn_wmma_f32_16x16x4_f32)
#define HAVE_WMMA_F32X4 1
#endif
#endif

// ---------------------------------------------------------------------------
// Kernel 1: dequantize weights -> f16, packed in WMMA B-operand register order.
// B (32x16): V0..7, lanes 0-15 hold K=0..7,16..23 (col n = lane), lanes 16-31 hold
// K=8..15,24..31. Per (kb, n): 32 f16 sequential; GEMM loads b128 x2 per lane.
// ---------------------------------------------------------------------------
__global__ __launch_bounds__(256) void pack_weights(
    const float* __restrict__ qw, const float* __restrict__ wsc,
    _Float16* __restrict__ wp) {
  int idx = blockIdx.x * 256 + threadIdx.x;
  if (idx >= KB * OUT_F) return;
  int kb = idx / OUT_F;
  int n  = idx - kb * OUT_F;   // consecutive threads -> consecutive n: coalesced
  _Float16 vals[32];
#pragma unroll
  for (int kk = 0; kk < 32; ++kk) {
    int k = kb * 32 + kk;
    float w = qw[(size_t)k * OUT_F + n] * wsc[(size_t)(k >> 6) * OUT_F + n];
    vals[kk] = (_Float16)w;
  }
  h8* dst = (h8*)(wp + (size_t)idx * 32);
#pragma unroll
  for (int j = 0; j < 4; ++j) {
    h8 v;
#pragma unroll
    for (int e = 0; e < 8; ++e) v[e] = vals[((j & 1) << 4) + ((j >> 1) << 3) + e];
    dst[j] = v;
  }
}

// ---------------------------------------------------------------------------
// Kernel 2: one 256-thread block per token (single HBM pass over x).
//  - stage x row in LDS, per-group absmax -> scale, fake-quant, store f16 in
//    WMMA A-operand register order: a_pack[mtile][kb][lane][16 halves]
//      lane = (m%16) + 16*((k>>3)&1),  half = (k&7) + 8*((k>>4)&1)
//  - same pass: t[token][r] = sum_k x[k] * proj_down[k][r]
// ---------------------------------------------------------------------------
__global__ __launch_bounds__(256) void quant_pack_lowrank(
    const float* __restrict__ x, const float* __restrict__ pd,
    _Float16* __restrict__ ap, float* __restrict__ t) {
  __shared__ float sx[IN_F];
  __shared__ float sscale[IN_F / GROUP];
  __shared__ float sred[8 * RANK];

  int token = blockIdx.x;
  int tid   = threadIdx.x;

  const float4* xr  = (const float4*)(x + (size_t)token * IN_F);
  float4*       sx4 = (float4*)sx;
  for (int i = tid; i < IN_F / 4; i += 256) sx4[i] = xr[i];
  __syncthreads();

  if (tid < IN_F / GROUP) {
    float m = 0.f;
    int base = tid * GROUP;
    for (int j = 0; j < GROUP; ++j) m = fmaxf(m, fabsf(sx[base + j]));
    sscale[tid] = fmaxf(m * (1.0f / 7.0f), 1e-6f);
  }
  __syncthreads();

  int mtile = token >> 4;
  int mlane = token & 15;
  for (int i = tid; i < IN_F / 4; i += 256) {
    int   k   = i * 4;                 // 4-chunk never straddles a group
    float s   = sscale[k >> 6];
    float inv = 1.0f / s;
    float4 v  = sx4[i];
    h4 q;
    q.x = (_Float16)(fminf(fmaxf(rintf(v.x * inv), -8.f), 7.f) * s);
    q.y = (_Float16)(fminf(fmaxf(rintf(v.y * inv), -8.f), 7.f) * s);
    q.z = (_Float16)(fminf(fmaxf(rintf(v.z * inv), -8.f), 7.f) * s);
    q.w = (_Float16)(fminf(fmaxf(rintf(v.w * inv), -8.f), 7.f) * s);
    int kk   = k & 31;
    int kb   = k >> 5;
    int lane = mlane + (((kk >> 3) & 1) << 4);
    int h    = (kk & 7) + (((kk >> 4) & 1) << 3);   // multiple of 4 -> 8B aligned
    _Float16* dst = ap + ((size_t)(mtile * KB + kb) * 32 + lane) * 16 + h;
    *(h4*)dst = q;
  }

  // low-rank projection t = x @ proj_down (uses UNquantized x, per reference)
  int r    = tid & 31;
  int part = tid >> 5;                 // 8 partitions of 720 k's
  const int KPART = IN_F / 8;
  float acc = 0.f;
  int k0 = part * KPART;
  for (int j = 0; j < KPART; ++j) {
    int k = k0 + j;
    acc = fmaf(sx[k], pd[(size_t)k * RANK + r], acc);  // lanes read contiguous 128B
  }
  sred[part * RANK + r] = acc;
  __syncthreads();
  if (tid < RANK) {
    float sum = 0.f;
#pragma unroll
    for (int p = 0; p < 8; ++p) sum += sred[p * RANK + tid];
    t[(size_t)token * RANK + tid] = sum;
  }
}

// ---------------------------------------------------------------------------
// Kernel 3: one wave32 per 32x32 output block (2x2 grid of 16x16 WMMA tiles).
// Register blocking: each A fragment feeds NT WMMAs, each B fragment feeds MT
// WMMAs -> operand traffic per WMMA drops from 2KB to 1KB per wave.
// 180 kb iters x 4 v_wmma_f32_16x16x32_f16, then 8x v_wmma_f32_16x16x4_f32
// per tile for the low-rank term, bias, NCHW store (lane's 8 accumulators are
// 8 consecutive tokens of one channel = contiguous floats in (N,C,H,W)).
// ---------------------------------------------------------------------------
__global__ __launch_bounds__(32) void gemm_wmma(
    const _Float16* __restrict__ ap, const _Float16* __restrict__ wp,
    const float* __restrict__ t, const float* __restrict__ pu,
    const float* __restrict__ bias, float* __restrict__ out) {
  int lane   = threadIdx.x;
  int l15    = lane & 15;
  int half   = lane >> 4;
  int mtile0 = blockIdx.x * MT;
  int ntile0 = blockIdx.y * NT;

  // a_pack stride per mtile = KB*32*16 halves; per kb = 32*16 halves
  const _Float16* a0 = ap + ((size_t)mtile0 * KB * 32 + lane) * 16;
  const _Float16* a1 = a0 + (size_t)KB * 32 * 16;
  // w_pack stride per n = 32 halves; per kb = OUT_F*32 halves
  const _Float16* w0 = wp + ((size_t)(ntile0 * 16 + l15)) * 32 + (size_t)half * 16;
  const _Float16* w1 = w0 + 16 * 32;

  v8f acc00 = {}, acc01 = {}, acc10 = {}, acc11 = {};
  for (int kb = 0; kb < KB; ++kb) {
    v16h A0 = *(const v16h*)a0;        // 2x global_load_b128 each, coalesced
    v16h A1 = *(const v16h*)a1;
    v16h B0 = *(const v16h*)w0;
    v16h B1 = *(const v16h*)w1;
    __builtin_prefetch(a0 + 8 * 32 * 16, 0, 3);  // A-stream, 8 kb ahead, near caches
    __builtin_prefetch(a1 + 8 * 32 * 16, 0, 3);
    acc00 = __builtin_amdgcn_wmma_f32_16x16x32_f16(false, A0, false, B0, (short)0, acc00, false, false);
    acc01 = __builtin_amdgcn_wmma_f32_16x16x32_f16(false, A0, false, B1, (short)0, acc01, false, false);
    acc10 = __builtin_amdgcn_wmma_f32_16x16x32_f16(false, A1, false, B0, (short)0, acc10, false, false);
    acc11 = __builtin_amdgcn_wmma_f32_16x16x32_f16(false, A1, false, B1, (short)0, acc11, false, false);
    a0 += 32 * 16;
    a1 += 32 * 16;
    w0 += OUT_F * 32;
    w1 += OUT_F * 32;
  }

  v8f accs[MT][NT] = {{acc00, acc01}, {acc10, acc11}};

#pragma unroll
  for (int i = 0; i < MT; ++i) {
    int mtile = mtile0 + i;
#pragma unroll
    for (int j = 0; j < NT; ++j) {
      int n = (ntile0 + j) * 16 + l15;
      v8f acc = accs[i][j];

      // Low-rank epilogue: acc += t_tile(16x32) @ proj_up(32x16)
#if defined(HAVE_WMMA_F32X4)
      const float* trow = t + ((size_t)(mtile * 16 + l15)) * RANK;  // A: m = lane&15
#pragma unroll
      for (int jj = 0; jj < 8; ++jj) {
        int k0 = jj * 4 + half * 2;    // A 16x4: V0=K(half*2), V1=K(half*2+1)
        v2f af, bf;
        af.x = trow[k0];
        af.y = trow[k0 + 1];
        bf.x = pu[(size_t)k0 * OUT_F + n];
        bf.y = pu[(size_t)(k0 + 1) * OUT_F + n];
        acc = __builtin_amdgcn_wmma_f32_16x16x4_f32(false, af, false, bf, (short)0, acc, false, false);
      }
#else
      {
        int mg = mtile * 16 + half * 8;
#pragma unroll
        for (int r = 0; r < 8; ++r) {
          const float* trow = t + (size_t)(mg + r) * RANK;
          float s = 0.f;
          for (int k = 0; k < RANK; ++k) s = fmaf(trow[k], pu[(size_t)k * OUT_F + n], s);
          acc[r] += s;
        }
      }
#endif

      float bv  = bias[n];
      int   mg0 = mtile * 16 + half * 8;  // 8 consecutive tokens, same batch
      int   b   = mg0 >> 12;              // token / 4096
      int   l   = mg0 & 4095;
      float* o  = out + ((size_t)(b * OUT_F + n)) * 4096 + l;
      float4 lo, hi;
      lo.x = acc[0] + bv; lo.y = acc[1] + bv; lo.z = acc[2] + bv; lo.w = acc[3] + bv;
      hi.x = acc[4] + bv; hi.y = acc[5] + bv; hi.z = acc[6] + bv; hi.w = acc[7] + bv;
      *(float4*)o       = lo;             // global_store_b128 x2, 32B aligned
      *(float4*)(o + 4) = hi;
    }
  }
}

// ---------------------------------------------------------------------------
// Launch. Workspace: a_pack 188.7MB | w_pack 7.4MB | t 2.1MB  (~198MB)
// ---------------------------------------------------------------------------
extern "C" void kernel_launch(void* const* d_in, const int* in_sizes, int n_in,
                              void* d_out, int out_size, void* d_ws, size_t ws_size,
                              hipStream_t stream) {
  const float* x    = (const float*)d_in[0];
  const float* qw   = (const float*)d_in[1];
  const float* wsc  = (const float*)d_in[2];
  const float* pd   = (const float*)d_in[3];
  const float* pu   = (const float*)d_in[4];
  const float* bias = (const float*)d_in[5];
  float*       out  = (float*)d_out;

  char* ws = (char*)d_ws;
  _Float16* a_pack = (_Float16*)ws;
  _Float16* w_pack = (_Float16*)(ws + (size_t)NTOK * IN_F * 2);
  float*    t      = (float*)(ws + (size_t)NTOK * IN_F * 2 + (size_t)IN_F * OUT_F * 2);

  pack_weights<<<(KB * OUT_F + 255) / 256, 256, 0, stream>>>(qw, wsc, w_pack);
  quant_pack_lowrank<<<NTOK, 256, 0, stream>>>(x, pd, a_pack, t);
  gemm_wmma<<<dim3(MTILES / MT, NTILES / NT), 32, 0, stream>>>(a_pack, w_pack, t, pu, bias, out);
}
